// GAT_Encoder_88776974008749
// MI455X (gfx1250) — matrix operand order repaired and checked by reading, not compiled, not run on hardware
//
#include <hip/hip_runtime.h>
#include <hip/hip_bf16.h>
#include <math.h>

typedef __attribute__((ext_vector_type(16))) _Float16 v16h;
typedef __attribute__((ext_vector_type(8)))  float    v8f;

#define NEG_SLOPE 0.2f
#define BN_EPS    1e-5f

// ---------------------------------------------------------------- utilities
__global__ void fill_f32(float* p, float v, int n) {
  int i = blockIdx.x * 256 + threadIdx.x;
  if (i < n) p[i] = v;
}

__device__ __forceinline__ void atomicMaxF(float* addr, float val) {
  unsigned int* ua = (unsigned int*)addr;
  unsigned int old = __float_as_uint(*addr);
  while (__uint_as_float(old) < val) {
    unsigned int assumed = old;
    old = atomicCAS(ua, assumed, __float_as_uint(val));
    if (old == assumed) break;
  }
}

// ------------------------------------------------- self-loop mean edge_attr
__global__ void deg_sum_kernel(const int* __restrict__ dst,
                               const float* __restrict__ ea,
                               float* __restrict__ loopsum,
                               float* __restrict__ deg,
                               int E, int ED) {
  int gid = blockIdx.x * 256 + threadIdx.x;
  if (gid >= E * ED) return;
  int e = gid / ED, d = gid % ED;
  int dn = dst[e];
  atomicAdd(loopsum + (size_t)dn * ED + d, ea[(size_t)e * ED + d]);
  if (d == 0) atomicAdd(deg + dn, 1.0f);
}

__global__ void loop_div_kernel(float* __restrict__ loopmean,
                                const float* __restrict__ deg, int N, int ED) {
  int gid = blockIdx.x * 256 + threadIdx.x;
  if (gid >= N * ED) return;
  loopmean[gid] /= fmaxf(deg[gid / ED], 1.0f);
}

// ---------------------------------- wE[d,h] = sum_f We[d, h*Fo+f]*a_e[h,f]
__global__ void we_prep_kernel(const float* __restrict__ We,
                               const float* __restrict__ ae,
                               float* __restrict__ wE, int ED, int H, int Fo) {
  int gid = blockIdx.x * 128 + threadIdx.x;
  if (gid >= ED * H) return;
  int d = gid / H, h = gid % H;
  float s = 0.f;
  for (int f = 0; f < Fo; ++f)
    s += We[(size_t)d * H * Fo + h * Fo + f] * ae[h * Fo + f];
  wE[gid] = s;
}

// --------------------------------------------- WMMA GEMM  C[M,Nc]=A[M,K]@B
// 4 waves per block; each wave computes one 16x16 tile. Requires:
//   M % 16 == 0, K % 32 == 0, Nc % 16 == 0  (true for all call sites here)
// f32 in memory, f16 multiply, f32 accumulate (v_wmma_f32_16x16x32_f16).
__global__ void wmma_gemm_f16f32(const float* __restrict__ A,
                                 const float* __restrict__ Bm,
                                 float* __restrict__ C,
                                 int M, int K, int Nc) {
  const int lane = threadIdx.x & 31;
  const int wave = threadIdx.x >> 5;
  const int tn = (blockIdx.x * 4 + wave) * 16;
  const int tm = blockIdx.y * 16;
  if (tn >= Nc) return;                 // wave-uniform
  const int hs  = lane >> 4;            // half-wave select
  const int l15 = lane & 15;
  const int col = tn + l15;
  const float* Arow = A + (size_t)(tm + l15) * K;
  v8f acc = {};
  for (int kk = 0; kk < K; kk += 32) {
    // A fragment: element i -> K = kk + hs*16 + i
    const float4 a0 = *(const float4*)(Arow + kk + hs * 16);
    const float4 a1 = *(const float4*)(Arow + kk + hs * 16 + 4);
    const float4 a2 = *(const float4*)(Arow + kk + hs * 16 + 8);
    const float4 a3 = *(const float4*)(Arow + kk + hs * 16 + 12);
    v16h a;
    a[0]  = (_Float16)a0.x; a[1]  = (_Float16)a0.y; a[2]  = (_Float16)a0.z; a[3]  = (_Float16)a0.w;
    a[4]  = (_Float16)a1.x; a[5]  = (_Float16)a1.y; a[6]  = (_Float16)a1.z; a[7]  = (_Float16)a1.w;
    a[8]  = (_Float16)a2.x; a[9]  = (_Float16)a2.y; a[10] = (_Float16)a2.z; a[11] = (_Float16)a2.w;
    a[12] = (_Float16)a3.x; a[13] = (_Float16)a3.y; a[14] = (_Float16)a3.z; a[15] = (_Float16)a3.w;
    // B fragment: element i -> K = kk + hs*16 + i, N = col (coalesced per i)
    v16h b;
    const int kbB = kk + hs * 16;
#pragma unroll
    for (int i = 0; i < 16; ++i)
      b[i] = (_Float16)Bm[(size_t)(kbB + i) * Nc + col];
    acc = __builtin_amdgcn_wmma_f32_16x16x32_f16(
        false, a, false, b, (short)0, acc, false, false);
  }
#pragma unroll
  for (int r = 0; r < 8; ++r)
    C[(size_t)(tm + r + hs * 8) * Nc + col] = acc[r];
}

// ---------------------------- per-node attention logits al_src/al_dst [N,H]
__global__ void al_kernel(const float* __restrict__ xs,
                          const float* __restrict__ as_,
                          const float* __restrict__ ad_,
                          float* __restrict__ als, float* __restrict__ ald,
                          int N, int H, int Fo) {
  int gid = blockIdx.x * 256 + threadIdx.x;
  if (gid >= N * H) return;
  int n = gid / H, h = gid % H;
  const float* row = xs + (size_t)n * H * Fo + h * Fo;
  float s = 0.f, d = 0.f;
  for (int f = 0; f < Fo; ++f) { s += row[f] * as_[h * Fo + f]; d += row[f] * ad_[h * Fo + f]; }
  als[gid] = s; ald[gid] = d;
}

// -------------------- per-edge raw attention + running segment max (atomic)
__global__ void alpha_kernel(const int* __restrict__ ei,
                             const float* __restrict__ ea,
                             const float* __restrict__ loopmean,
                             const float* __restrict__ wE,
                             const float* __restrict__ als,
                             const float* __restrict__ ald,
                             float* __restrict__ alp, float* __restrict__ mx,
                             int E, int N, int H, int ED) {
  int gid = blockIdx.x * 256 + threadIdx.x;
  int Et = E + N;
  if (gid >= Et * H) return;
  int e = gid / H, h = gid % H;
  int s, d;
  const float* earow;
  if (e < E) { s = ei[e]; d = ei[E + e]; earow = ea + (size_t)e * ED; }
  else       { s = d = e - E;            earow = loopmean + (size_t)(e - E) * ED; }
  float aE = 0.f;
  for (int dd = 0; dd < ED; ++dd) aE += earow[dd] * wE[dd * H + h];
  float v = als[s * H + h] + ald[d * H + h] + aE;
  v = (v > 0.f) ? v : NEG_SLOPE * v;
  alp[gid] = v;
  atomicMaxF(mx + d * H + h, v);
}

// ------------------------------------- exp(alpha - max) and segment sum
__global__ void ex_kernel(const int* __restrict__ ei,
                          float* __restrict__ alp,
                          const float* __restrict__ mx,
                          float* __restrict__ ssum,
                          int E, int N, int H) {
  int gid = blockIdx.x * 256 + threadIdx.x;
  int Et = E + N;
  if (gid >= Et * H) return;
  int e = gid / H, h = gid % H;
  int d = (e < E) ? ei[E + e] : (e - E);
  float ex = __expf(alp[gid] - mx[d * H + h]);
  alp[gid] = ex;
  atomicAdd(ssum + d * H + h, ex);
}

// --------------------- out[dst] += xs[src] * (ex/sum[dst]) per channel
__global__ void agg_kernel(const int* __restrict__ ei,
                           const float* __restrict__ ex,
                           const float* __restrict__ ssum,
                           const float* __restrict__ xs,
                           float* __restrict__ out,
                           int E, int N, int H, int Fo) {
  const int C = H * Fo;
  long long gid = (long long)blockIdx.x * 256 + threadIdx.x;
  long long total = (long long)(E + N) * C;
  if (gid >= total) return;
  int e = (int)(gid / C), c = (int)(gid % C);
  int h = c / Fo;
  int s, d;
  if (e < E) { s = ei[e]; d = ei[E + e]; } else { s = d = e - E; }
  float w = ex[(size_t)e * H + h] / ssum[d * H + h];
  atomicAdd(out + (size_t)d * C + c, xs[(size_t)s * C + c] * w);
}

// ------------------------------------------------ BatchNorm (training stats)
__global__ void bn_stats_kernel(const float* __restrict__ X,
                                float* __restrict__ mu, float* __restrict__ rs,
                                int Nn, int C) {
  int c = blockIdx.x;
  __shared__ float ss[256], sq[256];
  float a = 0.f, b = 0.f;
  for (int n = threadIdx.x; n < Nn; n += 256) {
    float v = X[(size_t)n * C + c];
    a += v; b += v * v;
  }
  ss[threadIdx.x] = a; sq[threadIdx.x] = b;
  __syncthreads();
  for (int st = 128; st > 0; st >>= 1) {
    if (threadIdx.x < st) { ss[threadIdx.x] += ss[threadIdx.x + st]; sq[threadIdx.x] += sq[threadIdx.x + st]; }
    __syncthreads();
  }
  if (threadIdx.x == 0) {
    float m = ss[0] / Nn;
    float v = sq[0] / Nn - m * m;
    mu[c] = m; rs[c] = rsqrtf(v + BN_EPS);
  }
}

__global__ void bn_relu_kernel(float* __restrict__ X,
                               const float* __restrict__ mu,
                               const float* __restrict__ rs,
                               const float* __restrict__ g,
                               const float* __restrict__ b,
                               int total, int C) {
  int gid = blockIdx.x * 256 + threadIdx.x;
  if (gid >= total) return;
  int c = gid % C;
  X[gid] = fmaxf(0.f, (X[gid] - mu[c]) * rs[c] * g[c] + b[c]);
}

// ------------------------------------------------------- attention pooling
__global__ void gate_kernel(const float* __restrict__ h2,
                            const float* __restrict__ Wg, const float* __restrict__ bg,
                            const int* __restrict__ batch,
                            float* __restrict__ gate, float* __restrict__ pm,
                            int N, int C) {
  int n = blockIdx.x * 256 + threadIdx.x;
  if (n >= N) return;
  float s = bg[0];
  for (int c = 0; c < C; ++c) s += h2[(size_t)n * C + c] * Wg[c];
  gate[n] = s;
  atomicMaxF(pm + batch[n], s);
}

__global__ void pool_ex_kernel(float* __restrict__ gate,
                               const float* __restrict__ pm,
                               float* __restrict__ ps,
                               const int* __restrict__ batch, int N) {
  int n = blockIdx.x * 256 + threadIdx.x;
  if (n >= N) return;
  int b = batch[n];
  float e = __expf(gate[n] - pm[b]);
  gate[n] = e;
  atomicAdd(ps + b, e);
}

__global__ void pool_sum_kernel(const float* __restrict__ h2,
                                const float* __restrict__ gate,
                                const float* __restrict__ ps,
                                const int* __restrict__ batch,
                                float* __restrict__ out, int N, int C) {
  int gid = blockIdx.x * 256 + threadIdx.x;
  if (gid >= N * C) return;
  int n = gid / C, c = gid % C;
  int b = batch[n];
  atomicAdd(out + (size_t)b * C + c, h2[(size_t)n * C + c] * gate[n] / ps[b]);
}

// ===========================================================================
extern "C" void kernel_launch(void* const* d_in, const int* in_sizes, int n_in,
                              void* d_out, int out_size, void* d_ws, size_t ws_size,
                              hipStream_t stream) {
  const int N = 30000, E = 480000, B = 64, ED = 32;
  const int Et = E + N;
  const int H1 = 4, H2 = 1, Fo = 64;
  const int C1 = H1 * Fo;   // 256
  const int C2 = H2 * Fo;   // 64
  const int FIN = 128;

  const float* x    = (const float*)d_in[0];
  const float* ea   = (const float*)d_in[1];
  const int*   ei   = (const int*)  d_in[2];   // [2,E] flat: src=ei[e], dst=ei[E+e]
  const int*   batch= (const int*)  d_in[3];
  const float* W1   = (const float*)d_in[4];
  const float* We1  = (const float*)d_in[5];
  const float* as1  = (const float*)d_in[6];
  const float* ad1  = (const float*)d_in[7];
  const float* ae1  = (const float*)d_in[8];
  // bias1 (d_in[9]) is a no-op through BatchNorm -> skipped
  const float* g1   = (const float*)d_in[10];
  const float* b1   = (const float*)d_in[11];
  const float* W2   = (const float*)d_in[12];
  const float* We2  = (const float*)d_in[13];
  const float* as2  = (const float*)d_in[14];
  const float* ad2  = (const float*)d_in[15];
  const float* ae2  = (const float*)d_in[16];
  // bias2 (d_in[17]) skipped for the same reason
  const float* g2   = (const float*)d_in[18];
  const float* b2   = (const float*)d_in[19];
  const float* Wg   = (const float*)d_in[20];
  const float* bg   = (const float*)d_in[21];
  float* out = (float*)d_out;

  // -------- workspace layout (floats, 64-elem aligned chunks)
  float* wsf = (float*)d_ws;
  size_t off = 0;
  auto alloc = [&](size_t n) { float* p = wsf + off; off += (n + 63) & ~(size_t)63; return p; };
  float* deg      = alloc(N);
  float* loopmean = alloc((size_t)N * ED);
  float* wE1      = alloc(ED * H1);
  float* wE2      = alloc(ED * H2);
  float* xs1      = alloc((size_t)N * C1);
  float* als1     = alloc((size_t)N * H1);
  float* ald1     = alloc((size_t)N * H1);
  float* m1       = alloc((size_t)N * H1);
  float* s1       = alloc((size_t)N * H1);
  float* alp1     = alloc((size_t)Et * H1);
  float* h1       = alloc((size_t)N * C1);
  float* mu1      = alloc(C1);
  float* rs1      = alloc(C1);
  float* xs2      = alloc((size_t)N * C2);
  float* als2     = alloc(N);
  float* ald2     = alloc(N);
  float* m2       = alloc(N);
  float* s2       = alloc(N);
  float* alp2     = alloc(Et);
  float* h2       = alloc((size_t)N * C2);
  float* mu2      = alloc(C2);
  float* rs2      = alloc(C2);
  float* gate     = alloc(N);
  float* pm       = alloc(B);
  float* ps       = alloc(B);
  (void)ws_size; (void)n_in; (void)in_sizes; (void)out_size;

  auto blk = [](long long n) { return dim3((unsigned)((n + 255) / 256)); };
  const dim3 T(256);
  const float NEGINF = -3.0e38f;

  // -------- init accumulators (ws is poisoned between runs)
  fill_f32<<<blk(N), T, 0, stream>>>(deg, 0.f, N);
  fill_f32<<<blk((long long)N * ED), T, 0, stream>>>(loopmean, 0.f, N * ED);
  fill_f32<<<blk((long long)N * H1), T, 0, stream>>>(m1, NEGINF, N * H1);
  fill_f32<<<blk((long long)N * H1), T, 0, stream>>>(s1, 0.f, N * H1);
  fill_f32<<<blk((long long)N * C1), T, 0, stream>>>(h1, 0.f, N * C1);
  fill_f32<<<blk(N), T, 0, stream>>>(m2, NEGINF, N);
  fill_f32<<<blk(N), T, 0, stream>>>(s2, 0.f, N);
  fill_f32<<<blk((long long)N * C2), T, 0, stream>>>(h2, 0.f, N * C2);
  fill_f32<<<blk(B), T, 0, stream>>>(pm, NEGINF, B);
  fill_f32<<<blk(B), T, 0, stream>>>(ps, 0.f, B);
  fill_f32<<<blk(B * C2), T, 0, stream>>>(out, 0.f, B * C2);

  // -------- self-loop mean edge_attr
  deg_sum_kernel<<<blk((long long)E * ED), T, 0, stream>>>(ei + E, ea, loopmean, deg, E, ED);
  loop_div_kernel<<<blk((long long)N * ED), T, 0, stream>>>(loopmean, deg, N, ED);

  // -------- folded edge-attention vectors (kills the 522 MB ef tensor)
  we_prep_kernel<<<1, 128, 0, stream>>>(We1, ae1, wE1, ED, H1, Fo);
  we_prep_kernel<<<1, 128, 0, stream>>>(We2, ae2, wE2, ED, H2, Fo);

  // ======== layer 1 ========
  wmma_gemm_f16f32<<<dim3((C1 / 16 + 3) / 4, N / 16), dim3(128), 0, stream>>>(x, W1, xs1, N, FIN, C1);
  al_kernel<<<blk((long long)N * H1), T, 0, stream>>>(xs1, as1, ad1, als1, ald1, N, H1, Fo);
  alpha_kernel<<<blk((long long)Et * H1), T, 0, stream>>>(ei, ea, loopmean, wE1, als1, ald1, alp1, m1, E, N, H1, ED);
  ex_kernel<<<blk((long long)Et * H1), T, 0, stream>>>(ei, alp1, m1, s1, E, N, H1);
  agg_kernel<<<blk((long long)Et * C1), T, 0, stream>>>(ei, alp1, s1, xs1, h1, E, N, H1, Fo);
  bn_stats_kernel<<<C1, 256, 0, stream>>>(h1, mu1, rs1, N, C1);
  bn_relu_kernel<<<blk((long long)N * C1), T, 0, stream>>>(h1, mu1, rs1, g1, b1, N * C1, C1);

  // ======== layer 2 ========
  wmma_gemm_f16f32<<<dim3((C2 / 16 + 3) / 4, N / 16), dim3(128), 0, stream>>>(h1, W2, xs2, N, C1, C2);
  al_kernel<<<blk(N), T, 0, stream>>>(xs2, as2, ad2, als2, ald2, N, H2, Fo);
  alpha_kernel<<<blk(Et), T, 0, stream>>>(ei, ea, loopmean, wE2, als2, ald2, alp2, m2, E, N, H2, ED);
  ex_kernel<<<blk(Et), T, 0, stream>>>(ei, alp2, m2, s2, E, N, H2);
  agg_kernel<<<blk((long long)Et * C2), T, 0, stream>>>(ei, alp2, s2, xs2, h2, E, N, H2, Fo);
  bn_stats_kernel<<<C2, 256, 0, stream>>>(h2, mu2, rs2, N, C2);
  bn_relu_kernel<<<blk((long long)N * C2), T, 0, stream>>>(h2, mu2, rs2, g2, b2, N * C2, C2);

  // ======== global attention pooling ========
  gate_kernel<<<blk(N), T, 0, stream>>>(h2, Wg, bg, batch, gate, pm, N, C2);
  pool_ex_kernel<<<blk(N), T, 0, stream>>>(gate, pm, ps, batch, N);
  pool_sum_kernel<<<blk((long long)N * C2), T, 0, stream>>>(h2, gate, ps, batch, out, N, C2);
}
